// Model_2534030704816
// MI455X (gfx1250) — compile-verified, matrix operand order checked by reading, not run on hardware
//
#include <hip/hip_runtime.h>
#include <hip/hip_bf16.h>
#include <hip/hip_fp16.h>

// ---------------------------------------------------------------------------
// CDNA5 (gfx1250): GEMMs via v_wmma_f32_16x16x32_f16.  A-operands staged to
// LDS with double-buffered global_load_async_to_lds_b128 (ASYNCcnt); each
// B fragment is reused across two WMMAs (32-row tiles) to halve B traffic.
// ---------------------------------------------------------------------------

typedef __attribute__((ext_vector_type(16))) _Float16 v16h;
typedef __attribute__((ext_vector_type(8)))  _Float16 v8h;
typedef __attribute__((ext_vector_type(8)))  float    v8f;

#define ACT_NONE 0
#define ACT_SIG  1
#define ACT_RELU 2

template <int ACT>
__device__ __forceinline__ float activate(float x) {
    if (ACT == ACT_SIG)  return 1.0f / (1.0f + __expf(-x));
    if (ACT == ACT_RELU) return x > 0.0f ? x : 0.0f;
    return x;
}

// Per ISA 7.12.2 (16-bit A/B frags): lane holds K = kb..kb+7 and kb+16..kb+23
// of its row (kb = (lane>>4)*8).  p points at row[kb]; works for global or LDS.
__device__ __forceinline__ v16h frag_load(const _Float16* p) {
    v8h c0 = *(const v8h*)(p);
    v8h c1 = *(const v8h*)(p + 16);
    v16h r;
#pragma unroll
    for (int i = 0; i < 8; ++i) { r[i] = c0[i]; r[i + 8] = c1[i]; }
    return r;
}

__device__ __forceinline__ v8f wmma_f16(v16h a, v16h b, v8f c) {
    return __builtin_amdgcn_wmma_f32_16x16x32_f16(false, a, false, b,
                                                  (short)0, c, false, false);
}

__device__ __forceinline__ void async_b128(unsigned lds_off, const _Float16* g) {
    asm volatile("global_load_async_to_lds_b128 %0, %1, off"
                 :: "v"(lds_off), "v"(g) : "memory");
}

// ---------------------------------------------------------------------------
// Async-staged GEMM:  C[M,256] = act(A[M,K] * W + bias), Bt = W^T [256,K] f16.
// Block = 128 thr = 4 waves over a 32-row x 256-col tile; wave w owns cols
// [64w,64w+64) for both 16-row subtiles (B frags reused x2).  A staged in
// LDS per 64-deep K chunk, double-buffered async copies.
// Requires K % 64 == 0, M % 32 == 0.  grid.x = M/32.
// ---------------------------------------------------------------------------
template <int ACT>
__global__ __launch_bounds__(128)
void gemm_wmma_async(const _Float16* __restrict__ A, const _Float16* __restrict__ Bt,
                     const float* __restrict__ bias, float* __restrict__ Cf,
                     _Float16* __restrict__ Ch, int M, int N, int K) {
    __shared__ _Float16 abuf[2 * 32 * 64];  // double-buffered 32x64 chunks (8 KB)
    int t    = threadIdx.x;
    int lane = t & 31;
    int wav  = t >> 5;                      // n quadrant
    int mt   = blockIdx.x;                  // 32-row tile
    int half = lane >> 4, l15 = lane & 15, kb = half * 8;

    const _Float16* brow = Bt + (size_t)(wav * 64 + l15) * K + kb;

    // staging: 32 rows x 8 x 16B chunks = 256 chunks; thread t does 2 of them
    int r0 = t >> 3, c0 = (t & 7) * 8;      // rows 0..15 / halves
    const _Float16* ag0 = A + (size_t)(mt * 32 + r0) * K + c0;
    const _Float16* ag1 = A + (size_t)(mt * 32 + r0 + 16) * K + c0;
    unsigned lb0 = (unsigned)(size_t)&abuf[r0 * 64 + c0];
    unsigned lb1 = (unsigned)(size_t)&abuf[(r0 + 16) * 64 + c0];

    v8f acc[2][4];
#pragma unroll
    for (int i = 0; i < 2; ++i)
#pragma unroll
        for (int j = 0; j < 4; ++j) acc[i][j] = {};

    // prologue: chunk 0 -> buffer 0
    async_b128(lb0, ag0);
    async_b128(lb1, ag1);

    int nc = K >> 6;
    for (int c = 0; c < nc; ++c) {
        unsigned cur = (unsigned)(c & 1);
        if (c + 1 < nc) {
            unsigned nxt = cur ^ 1u;
            async_b128(lb0 + nxt * 4096u, ag0 + (c + 1) * 64);
            async_b128(lb1 + nxt * 4096u, ag1 + (c + 1) * 64);
            asm volatile("s_wait_asynccnt 0x2" ::: "memory");   // chunk c landed
        } else {
            asm volatile("s_wait_asynccnt 0x0" ::: "memory");
        }
        __syncthreads();
        const _Float16* ab = abuf + cur * 2048u;                // halves
#pragma unroll
        for (int s = 0; s < 2; ++s) {
            int kk = c * 64 + s * 32;
            v16h a0 = frag_load(ab + l15 * 64 + s * 32 + kb);
            v16h a1 = frag_load(ab + (l15 + 16) * 64 + s * 32 + kb);
#pragma unroll
            for (int j = 0; j < 4; ++j) {
                v16h b = frag_load(brow + kk + (size_t)(j * 16) * K);
                acc[0][j] = wmma_f16(a0, b, acc[0][j]);
                acc[1][j] = wmma_f16(a1, b, acc[1][j]);
            }
        }
        __syncthreads();                    // guard before buffer reuse
    }
    int rowb = mt * 32 + half * 8;
#pragma unroll
    for (int i = 0; i < 2; ++i) {
#pragma unroll
        for (int j = 0; j < 4; ++j) {
            int col = wav * 64 + j * 16 + l15;
            float bv = bias[col];
#pragma unroll
            for (int r = 0; r < 8; ++r) {
                float v = activate<ACT>(acc[i][j][r] + bv);
                size_t o = (size_t)(rowb + i * 16 + r) * N + col;
                if (Cf) Cf[o] = v;
                if (Ch) Ch[o] = (_Float16)v;
            }
        }
    }
}

// ---------------------------------------------------------------------------
// Direct-global GEMM (kept for K=32 encoder layer 1).
// One wave per 16x64 tile over (M/16)*(N/64) tiles.
// ---------------------------------------------------------------------------
template <int ACT>
__global__ __launch_bounds__(128)
void gemm_wmma(const _Float16* __restrict__ A, const _Float16* __restrict__ Bt,
               const float* __restrict__ bias, float* __restrict__ Cf,
               _Float16* __restrict__ Ch, int M, int N, int K) {
    int wave = (int)((blockIdx.x * blockDim.x + threadIdx.x) >> 5);
    int lane = threadIdx.x & 31;
    int mtiles = M >> 4;
    if (wave >= mtiles * (N >> 6)) return;
    int mt = wave % mtiles, nq = wave / mtiles;
    int half = lane >> 4, l15 = lane & 15, kb = half * 8;

    const _Float16* arow = A + (size_t)(mt * 16 + l15) * K + kb;
    const _Float16* brow = Bt + (size_t)(nq * 64 + l15) * K + kb;
    v8f acc0 = {}, acc1 = {}, acc2 = {}, acc3 = {};
    for (int kk = 0; kk < K; kk += 32) {
        v16h a  = frag_load(arow + kk);
        v16h b0 = frag_load(brow + kk);
        v16h b1 = frag_load(brow + kk + (size_t)16 * K);
        v16h b2 = frag_load(brow + kk + (size_t)32 * K);
        v16h b3 = frag_load(brow + kk + (size_t)48 * K);
        acc0 = wmma_f16(a, b0, acc0);
        acc1 = wmma_f16(a, b1, acc1);
        acc2 = wmma_f16(a, b2, acc2);
        acc3 = wmma_f16(a, b3, acc3);
    }
    int rowb = mt * 16 + half * 8;
    v8f accs[4] = {acc0, acc1, acc2, acc3};
#pragma unroll
    for (int j = 0; j < 4; ++j) {
        int col = nq * 64 + j * 16 + l15;
        float bv = bias[col];
#pragma unroll
        for (int r = 0; r < 8; ++r) {
            float v = activate<ACT>(accs[j][r] + bv);
            size_t o = (size_t)(rowb + r) * N + col;
            if (Cf) Cf[o] = v;
            if (Ch) Ch[o] = (_Float16)v;
        }
    }
}

// ---------------------------------------------------------------------------
// Conv1 (k=10,s=10, 3->256 on 50x50) as GEMM with COOPERATIVE im2col in LDS.
// Block = 4 waves share one 32-row tile (grid.x = 3200); the 32x320 f16
// A-tile is materialized once, then the K loop is ds_load + WMMA with B
// frags reused across the two 16-row subtiles.  sigmoid -> f16 [102400,256].
// ---------------------------------------------------------------------------
__global__ __launch_bounds__(128)
void conv1_wmma(const float* __restrict__ X, const _Float16* __restrict__ Bt,
                const float* __restrict__ bias, _Float16* __restrict__ Ch) {
    const int K = 320, N = 256;
    const int LDK = 336;                      // 672 B row pitch (16B aligned)
    __shared__ _Float16 abuf[32 * LDK];       // 21 KB
    int t = threadIdx.x, lane = t & 31, wav = t >> 5;
    int mt = blockIdx.x;

    for (int idx = t; idx < 32 * 320; idx += 128) {
        int row = idx / 320, k = idx - row * 320;
        int m = mt * 32 + row;
        int b = m / 25, sp = m - b * 25;
        int oy = sp / 5, ox = sp - oy * 5;
        float v = 0.0f;
        if (k < 300) {
            int c = k / 100, rem = k - c * 100;
            int ky = rem / 10, kx = rem - ky * 10;
            v = X[((size_t)b * 3 + c) * 2500 + (oy * 10 + ky) * 50 + (ox * 10 + kx)];
        }
        abuf[row * LDK + k] = (_Float16)v;
    }
    __syncthreads();

    int half = lane >> 4, l15 = lane & 15, kb = half * 8;
    const _Float16* brow = Bt + (size_t)(wav * 64 + l15) * K + kb;
    const _Float16* a0p = &abuf[l15 * LDK + kb];
    const _Float16* a1p = &abuf[(l15 + 16) * LDK + kb];
    v8f acc[2][4];
#pragma unroll
    for (int i = 0; i < 2; ++i)
#pragma unroll
        for (int j = 0; j < 4; ++j) acc[i][j] = {};
#pragma unroll
    for (int kk = 0; kk < K; kk += 32) {
        v16h a0 = frag_load(a0p + kk);
        v16h a1 = frag_load(a1p + kk);
#pragma unroll
        for (int j = 0; j < 4; ++j) {
            v16h b = frag_load(brow + kk + (size_t)(j * 16) * K);
            acc[0][j] = wmma_f16(a0, b, acc[0][j]);
            acc[1][j] = wmma_f16(a1, b, acc[1][j]);
        }
    }
    int rowb = mt * 32 + half * 8;
#pragma unroll
    for (int i = 0; i < 2; ++i) {
#pragma unroll
        for (int j = 0; j < 4; ++j) {
            int col = wav * 64 + j * 16 + l15;
            float bv = bias[col];
#pragma unroll
            for (int r = 0; r < 8; ++r) {
                float v = activate<ACT_SIG>(acc[i][j][r] + bv);
                Ch[(size_t)(rowb + i * 16 + r) * N + col] = (_Float16)v;
            }
        }
    }
}

// ---------------------------------------------------------------------------
// Edge GEMM: A row r (r = b*20+p) is cat(z[b,src[p]], z[b,tgt[p]]) (K=512),
// gathered on the fly from f16 z [B*5,256] (L2-resident).  32-row tiles,
// B frags reused x2.  relu -> f16 out [81920,256].
// ---------------------------------------------------------------------------
__constant__ int c_src[20] = {0,0,0,0, 1,1,1,1, 2,2,2,2, 3,3,3,3, 4,4,4,4};
__constant__ int c_tgt[20] = {1,2,3,4, 0,2,3,4, 0,1,3,4, 0,1,2,4, 0,1,2,3};

__global__ __launch_bounds__(128)
void edge_gemm(const _Float16* __restrict__ Z, const _Float16* __restrict__ Bt,
               const float* __restrict__ bias, _Float16* __restrict__ Ch) {
    const int K = 512, N = 256, MT = 2560;  // 81920/32
    int wave = (int)((blockIdx.x * blockDim.x + threadIdx.x) >> 5);
    int lane = threadIdx.x & 31;
    if (wave >= MT * (N >> 6)) return;
    int mt = wave % MT, nq = wave / MT;
    int half = lane >> 4, l15 = lane & 15, kb = half * 8;

    int r0 = mt * 32 + l15, r1 = r0 + 16;
    int bi0 = r0 / 20, p0 = r0 - bi0 * 20;
    int bi1 = r1 / 20, p1 = r1 - bi1 * 20;
    const _Float16* zs0 = Z + (size_t)(bi0 * 5 + c_src[p0]) * 256 + kb;
    const _Float16* zt0 = Z + (size_t)(bi0 * 5 + c_tgt[p0]) * 256 + kb;
    const _Float16* zs1 = Z + (size_t)(bi1 * 5 + c_src[p1]) * 256 + kb;
    const _Float16* zt1 = Z + (size_t)(bi1 * 5 + c_tgt[p1]) * 256 + kb;
    const _Float16* brow = Bt + (size_t)(nq * 64 + l15) * K + kb;

    v8f acc[2][4];
#pragma unroll
    for (int i = 0; i < 2; ++i)
#pragma unroll
        for (int j = 0; j < 4; ++j) acc[i][j] = {};
    for (int kk = 0; kk < K; kk += 32) {
        v16h a0 = (kk < 256) ? frag_load(zs0 + kk) : frag_load(zt0 + (kk - 256));
        v16h a1 = (kk < 256) ? frag_load(zs1 + kk) : frag_load(zt1 + (kk - 256));
#pragma unroll
        for (int j = 0; j < 4; ++j) {
            v16h b = frag_load(brow + kk + (size_t)(j * 16) * K);
            acc[0][j] = wmma_f16(a0, b, acc[0][j]);
            acc[1][j] = wmma_f16(a1, b, acc[1][j]);
        }
    }
    int rowb = mt * 32 + half * 8;
#pragma unroll
    for (int i = 0; i < 2; ++i) {
#pragma unroll
        for (int j = 0; j < 4; ++j) {
            int col = nq * 64 + j * 16 + l15;
            float bv = bias[col];
#pragma unroll
            for (int rr = 0; rr < 8; ++rr) {
                float v = activate<ACT_RELU>(acc[i][j][rr] + bv);
                Ch[(size_t)(rowb + i * 16 + rr) * N + col] = (_Float16)v;
            }
        }
    }
}

// ---------------------------------------------------------------------------
// Weight conversion: W (f32, [K,N] or [N,K]) -> Bt (f16, [N,Kp]) zero-padded.
// ---------------------------------------------------------------------------
__global__ void wconv_kernel(const float* __restrict__ W, _Float16* __restrict__ Bt,
                             int K, int N, int Kp, int already_t) {
    int t = blockIdx.x * blockDim.x + threadIdx.x;
    if (t >= N * Kp) return;
    int n = t / Kp, k = t - n * Kp;
    float v = 0.0f;
    if (k < K) v = already_t ? W[(size_t)n * K + k] : W[(size_t)k * N + n];
    Bt[t] = (_Float16)v;
}

// ---------------------------------------------------------------------------
// 1x1 conv (256->5) + build padded encoder input  EncA f16 [B*5, 32].
// ---------------------------------------------------------------------------
__global__ void conv2_encA(const _Float16* __restrict__ H1, const float* __restrict__ W2,
                           const float* __restrict__ B2, _Float16* __restrict__ EncA) {
    int t = blockIdx.x * blockDim.x + threadIdx.x;
    if (t >= 20480 * 32) return;
    int row = t >> 5, col = t & 31;
    _Float16 out = (_Float16)0.0f;
    if (col < 25) {
        int b = row / 5, j = row - b * 5;
        const _Float16* h = H1 + (size_t)(b * 25 + col) * 256;
        const float* w = W2 + j * 256;
        float acc = B2[j];
        for (int c = 0; c < 256; ++c) acc = fmaf((float)h[c], w[c], acc);
        out = (_Float16)acc;
    }
    EncA[t] = out;
}

// ---------------------------------------------------------------------------
// Row LayerNorm over 256 (biased var, eps=1e-5): f32 in -> f16 out.
// One wave per row, wave32 shuffle reduction.
// ---------------------------------------------------------------------------
__global__ __launch_bounds__(128)
void ln_kernel(const float* __restrict__ X, const float* __restrict__ g,
               const float* __restrict__ b, _Float16* __restrict__ Y, int rows) {
    int wave = (int)((blockIdx.x * blockDim.x + threadIdx.x) >> 5);
    int lane = threadIdx.x & 31;
    if (wave >= rows) return;
    const float* x = X + (size_t)wave * 256;
    float v[8];
    float s = 0.0f;
#pragma unroll
    for (int i = 0; i < 8; ++i) { v[i] = x[lane + 32 * i]; s += v[i]; }
#pragma unroll
    for (int o = 16; o > 0; o >>= 1) s += __shfl_xor(s, o, 32);
    float mean = s * (1.0f / 256.0f);
    float q = 0.0f;
#pragma unroll
    for (int i = 0; i < 8; ++i) { float d = v[i] - mean; q += d * d; }
#pragma unroll
    for (int o = 16; o > 0; o >>= 1) q += __shfl_xor(q, o, 32);
    float inv = rsqrtf(q * (1.0f / 256.0f) + 1e-5f);
    _Float16* y = Y + (size_t)wave * 256;
#pragma unroll
    for (int i = 0; i < 8; ++i) {
        int c = lane + 32 * i;
        y[c] = (_Float16)((v[i] - mean) * inv * g[c] + b[c]);
    }
}

// ---------------------------------------------------------------------------
// Node-MLP input  NA f16 [B*5, 768] = [ z | act_emb[action] | edge_sum ].
// ---------------------------------------------------------------------------
__global__ void build_nodeA(const _Float16* __restrict__ Z, const float* __restrict__ AE,
                            const int* __restrict__ act, const float* __restrict__ Ef,
                            _Float16* __restrict__ NA) {
    int t = blockIdx.x * blockDim.x + threadIdx.x;
    if (t >= 20480 * 768) return;
    int row = t / 768, k = t - row * 768;
    int b = row / 5, i = row - b * 5;
    _Float16 v;
    if (k < 256) {
        v = Z[(size_t)row * 256 + k];
    } else if (k < 512) {
        v = (_Float16)AE[(size_t)act[b] * 256 + (k - 256)];
    } else {
        const float* e = Ef + (size_t)(b * 20 + i * 4) * 256 + (k - 512);
        v = (_Float16)(e[0] + e[256] + e[512] + e[768]);
    }
    NA[t] = v;
}

// ---------------------------------------------------------------------------
// Host orchestration
// ---------------------------------------------------------------------------
extern "C" void kernel_launch(void* const* d_in, const int* in_sizes, int n_in,
                              void* d_out, int out_size, void* d_ws, size_t ws_size,
                              hipStream_t stream) {
    const float* obs      = (const float*)d_in[0];
    const float* nobs     = (const float*)d_in[1];
    const int*   action   = (const int*)d_in[2];
    const float* ext_w1   = (const float*)d_in[3];
    const float* ext_b1   = (const float*)d_in[4];
    const float* ext_w2   = (const float*)d_in[5];
    const float* ext_b2   = (const float*)d_in[6];
    const float* enc_w1   = (const float*)d_in[7];
    const float* enc_b1   = (const float*)d_in[8];
    const float* enc_w2   = (const float*)d_in[9];
    const float* enc_b2   = (const float*)d_in[10];
    const float* enc_ln_g = (const float*)d_in[11];
    const float* enc_ln_b = (const float*)d_in[12];
    const float* enc_w3   = (const float*)d_in[13];
    const float* enc_b3   = (const float*)d_in[14];
    const float* act_emb  = (const float*)d_in[15];
    const float* edge_w1  = (const float*)d_in[16];
    const float* edge_b1  = (const float*)d_in[17];
    const float* edge_w2  = (const float*)d_in[18];
    const float* edge_b2  = (const float*)d_in[19];
    const float* edge_ln_g= (const float*)d_in[20];
    const float* edge_ln_b= (const float*)d_in[21];
    const float* edge_w3  = (const float*)d_in[22];
    const float* edge_b3  = (const float*)d_in[23];
    const float* node_w1  = (const float*)d_in[24];
    const float* node_b1  = (const float*)d_in[25];
    const float* node_w2  = (const float*)d_in[26];
    const float* node_b2  = (const float*)d_in[27];
    const float* node_ln_g= (const float*)d_in[28];
    const float* node_ln_b= (const float*)d_in[29];
    const float* node_w3  = (const float*)d_in[30];
    const float* node_b3  = (const float*)d_in[31];
    (void)in_sizes; (void)n_in; (void)out_size; (void)ws_size;

    float* out = (float*)d_out;
    const size_t ZSZ = (size_t)4096 * 5 * 256;  // 5,242,880
    float* out_obsz = out;
    float* out_node = out + ZSZ;
    float* out_nobz = out + 2 * ZSZ;

    // -------- workspace bump allocator (lifetime-based aliasing) ----------
    char* ws = (char*)d_ws;
    size_t off = 0;
    auto alloc = [&](size_t bytes) -> void* {
        void* p = ws + off;
        off += (bytes + 255) & ~(size_t)255;
        return p;
    };
    _Float16* wc1t = (_Float16*)alloc((size_t)256 * 320 * 2);
    _Float16* we1t = (_Float16*)alloc((size_t)256 * 32 * 2);
    _Float16* we2t = (_Float16*)alloc((size_t)256 * 256 * 2);
    _Float16* we3t = (_Float16*)alloc((size_t)256 * 256 * 2);
    _Float16* wg1t = (_Float16*)alloc((size_t)256 * 512 * 2);
    _Float16* wg2t = (_Float16*)alloc((size_t)256 * 256 * 2);
    _Float16* wg3t = (_Float16*)alloc((size_t)256 * 256 * 2);
    _Float16* wn1t = (_Float16*)alloc((size_t)256 * 768 * 2);
    _Float16* wn2t = (_Float16*)alloc((size_t)256 * 256 * 2);
    _Float16* wn3t = (_Float16*)alloc((size_t)256 * 256 * 2);
    char* arenaA = (char*)alloc((size_t)102400 * 256 * 2);  // H1h -> Gh -> NA
    _Float16* EncA = (_Float16*)alloc((size_t)20480 * 32 * 2);
    _Float16* T1h  = (_Float16*)alloc((size_t)20480 * 256 * 2);
    float*    T2f  = (float*)alloc((size_t)20480 * 256 * 4);
    _Float16* T2h  = (_Float16*)alloc((size_t)20480 * 256 * 2);
    _Float16* Zh   = (_Float16*)alloc((size_t)20480 * 256 * 2);
    char* arenaB = (char*)alloc((size_t)81920 * 256 * 4);   // G2f -> Ef
    _Float16* G2h  = (_Float16*)alloc((size_t)81920 * 256 * 2);

    _Float16* H1h = (_Float16*)arenaA;
    _Float16* Gh  = (_Float16*)arenaA;
    _Float16* NA  = (_Float16*)arenaA;
    float*    G2f = (float*)arenaB;
    float*    Ef  = (float*)arenaB;

    // -------- weight conversion ----------
    auto wc = [&](const float* W, _Float16* Bt, int K, int N, int Kp, int tr) {
        int n = N * Kp;
        wconv_kernel<<<(n + 255) / 256, 256, 0, stream>>>(W, Bt, K, N, Kp, tr);
    };
    wc(ext_w1, wc1t, 300, 256, 320, 1);
    wc(enc_w1, we1t, 25, 256, 32, 0);
    wc(enc_w2, we2t, 256, 256, 256, 0);
    wc(enc_w3, we3t, 256, 256, 256, 0);
    wc(edge_w1, wg1t, 512, 256, 512, 0);
    wc(edge_w2, wg2t, 256, 256, 256, 0);
    wc(edge_w3, wg3t, 256, 256, 256, 0);
    wc(node_w1, wn1t, 768, 256, 768, 0);
    wc(node_w2, wn2t, 256, 256, 256, 0);
    wc(node_w3, wn3t, 256, 256, 256, 0);

    auto tblocks = [](int M, int N) { return ((M / 16) * (N / 64) + 3) / 4; };

    // -------- extract + encode (obs, then next_obs) -----------------------
    const float* imgs[2]  = {obs, nobs};
    float*       zouts[2] = {out_obsz, out_nobz};
    for (int s = 0; s < 2; ++s) {
        conv1_wmma<<<3200, 128, 0, stream>>>(imgs[s], wc1t, ext_b1, H1h);
        conv2_encA<<<(20480 * 32 + 255) / 256, 256, 0, stream>>>(H1h, ext_w2, ext_b2, EncA);
        gemm_wmma<ACT_SIG><<<tblocks(20480, 256), 128, 0, stream>>>(
            EncA, we1t, enc_b1, nullptr, T1h, 20480, 256, 32);
        gemm_wmma_async<ACT_SIG><<<20480 / 32, 128, 0, stream>>>(
            T1h, we2t, enc_b2, T2f, nullptr, 20480, 256, 256);
        ln_kernel<<<(20480 + 3) / 4, 128, 0, stream>>>(T2f, enc_ln_g, enc_ln_b, T2h, 20480);
        gemm_wmma_async<ACT_NONE><<<20480 / 32, 128, 0, stream>>>(
            T2h, we3t, enc_b3, zouts[s], (s == 0) ? Zh : nullptr, 20480, 256, 256);
    }

    // -------- edge MLP (gather fused into GEMM) ---------------------------
    edge_gemm<<<(2560 * 4) / 4, 128, 0, stream>>>(Zh, wg1t, edge_b1, Gh);
    gemm_wmma_async<ACT_NONE><<<81920 / 32, 128, 0, stream>>>(
        Gh, wg2t, edge_b2, G2f, nullptr, 81920, 256, 256);
    ln_kernel<<<(81920 + 3) / 4, 128, 0, stream>>>(G2f, edge_ln_g, edge_ln_b, G2h, 81920);
    gemm_wmma_async<ACT_NONE><<<81920 / 32, 128, 0, stream>>>(
        G2h, wg3t, edge_b3, Ef, nullptr, 81920, 256, 256);

    // -------- node MLP ----------------------------------------------------
    build_nodeA<<<(20480 * 768 + 255) / 256, 256, 0, stream>>>(Zh, act_emb, action, Ef, NA);
    gemm_wmma_async<ACT_RELU><<<20480 / 32, 128, 0, stream>>>(
        NA, wn1t, node_b1, nullptr, T1h, 20480, 256, 768);
    gemm_wmma_async<ACT_NONE><<<20480 / 32, 128, 0, stream>>>(
        T1h, wn2t, node_b2, T2f, nullptr, 20480, 256, 256);
    ln_kernel<<<(20480 + 3) / 4, 128, 0, stream>>>(T2f, node_ln_g, node_ln_b, T2h, 20480);
    gemm_wmma_async<ACT_NONE><<<20480 / 32, 128, 0, stream>>>(
        T2h, wn3t, node_b3, out_node, nullptr, 20480, 256, 256);
}
